// RewardAwareAttention_86320252715351
// MI455X (gfx1250) — matrix operand order repaired and checked by reading, not compiled, not run on hardware
//
#include <hip/hip_runtime.h>
#include <hip/hip_bf16.h>
#include <stdint.h>

#define HIDDEN  2048
#define NHEADS  16
#define HEADDIM 128
#define SEQ     2048
#define BATCH   2

typedef __attribute__((ext_vector_type(16))) __bf16 bf16x16;
typedef __attribute__((ext_vector_type(8)))  float  v8f;

struct FragAB { union { bf16x16 v; unsigned int u[8]; }; };
struct FragC  { union { v8f v; float f[8]; }; };

__device__ __forceinline__ unsigned short f2bf(float x) {
  unsigned int u = __float_as_uint(x);
  u += 0x7fffu + ((u >> 16) & 1u);        // round-to-nearest-even
  return (unsigned short)(u >> 16);
}

// ---- CDNA5 async global->LDS copy (ASYNCcnt-tracked), with safe fallback ----
#if defined(__AMDGCN__) && __has_builtin(__builtin_amdgcn_global_load_async_to_lds_b128) && \
    __has_builtin(__builtin_amdgcn_s_wait_asynccnt)
#define HAVE_ASYNC_LDS 1
typedef __attribute__((__vector_size__(4 * sizeof(int)))) int v4i;
typedef __attribute__((address_space(1))) v4i as1_v4i;
typedef __attribute__((address_space(3))) v4i as3_v4i;
#endif

__device__ __forceinline__ void copy16_g2l(const void* g, void* l) {
#ifdef HAVE_ASYNC_LDS
  __builtin_amdgcn_global_load_async_to_lds_b128((as1_v4i*)g, (as3_v4i*)l, 0, 0);
#else
  *(uint4*)l = *(const uint4*)g;
#endif
}
__device__ __forceinline__ void wait_async_keep4() {
#ifdef HAVE_ASYNC_LDS
  __builtin_amdgcn_s_wait_asynccnt(4);   // oldest 4 (current tile) complete
#endif
}
__device__ __forceinline__ void wait_async_all() {
#ifdef HAVE_ASYNC_LDS
  __builtin_amdgcn_s_wait_asynccnt(0);
#endif
}

// A fragment: 16x32 bf16 tile, row-major, lda in elements (even).
__device__ __forceinline__ void load_frag_a(FragAB& f, const unsigned short* base, int lda, int lane) {
  const int half = (lane >> 4) & 1;
  const int m = lane & 15;
  const unsigned short* p = base + m * lda + half * 16;
  const unsigned int* p0 = reinterpret_cast<const unsigned int*>(p);
#pragma unroll
  for (int i = 0; i < 8; ++i) f.u[i] = p0[i];
}

// B fragment: 32x16 bf16 tile stored transposed as Bt[n][k], ldb in elements (even).
__device__ __forceinline__ void load_frag_b(FragAB& f, const unsigned short* base, int ldb, int lane) {
  const int half = (lane >> 4) & 1;
  const int n = lane & 15;
  const unsigned int* p = reinterpret_cast<const unsigned int*>(base + n * ldb + half * 16);
#pragma unroll
  for (int i = 0; i < 8; ++i) f.u[i] = p[i];
}

__device__ __forceinline__ v8f wmma_bf16(bf16x16 a, bf16x16 b, v8f c) {
  return __builtin_amdgcn_wmma_f32_16x16x32_bf16(false, a, false, b, (short)0, c, false, false);
}

// ---------------------------------------------------------------------------
// reward bias constants: bias[b,h,q] = rewards[b,q]*c1[h] + c0[h]
// ---------------------------------------------------------------------------
__global__ void reward_consts_kernel(const float* __restrict__ Wr, const float* __restrict__ br,
                                     const float* __restrict__ Wa, const float* __restrict__ ba,
                                     float* __restrict__ c01) {
  __shared__ float r0[256], r1[256];
  const int h = blockIdx.x, tid = threadIdx.x;
  float s1 = 0.f, s0 = 0.f;
  for (int j = tid; j < HIDDEN; j += 256) {
    float wa = Wa[(size_t)j * NHEADS + h];
    s1 += Wr[j] * wa;
    s0 += br[j] * wa;
  }
  r1[tid] = s1; r0[tid] = s0;
  __syncthreads();
  for (int step = 128; step > 0; step >>= 1) {
    if (tid < step) { r1[tid] += r1[tid + step]; r0[tid] += r0[tid + step]; }
    __syncthreads();
  }
  if (tid == 0) { c01[h] = r1[0]; c01[NHEADS + h] = r0[0] + ba[h]; }
}

// ---------------------------------------------------------------------------
// Wt[n][k] = bf16(W[k][n]); 2048x2048
// ---------------------------------------------------------------------------
__global__ void transpose_w_kernel(const float* __restrict__ W, unsigned short* __restrict__ Wt) {
  __shared__ float tile[32][33];
  const int nb = blockIdx.x * 32, kb = blockIdx.y * 32;
  const int tx = threadIdx.x, ty = threadIdx.y;
#pragma unroll
  for (int i = 0; i < 32; i += 8)
    tile[ty + i][tx] = W[(size_t)(kb + ty + i) * HIDDEN + nb + tx];
  __syncthreads();
#pragma unroll
  for (int i = 0; i < 32; i += 8)
    Wt[(size_t)(nb + ty + i) * HIDDEN + kb + tx] = f2bf(tile[tx][ty + i]);
}

// ---------------------------------------------------------------------------
// streaming f32 -> bf16 convert (4 elems/thread, float4 loads)
// ---------------------------------------------------------------------------
__global__ void cvt_bf16_kernel(const float* __restrict__ X, unsigned short* __restrict__ Y) {
  const int i = blockIdx.x * 256 + threadIdx.x;
  const float4 v = ((const float4*)X)[i];
  uint2 o;
  o.x = (unsigned int)f2bf(v.x) | ((unsigned int)f2bf(v.y) << 16);
  o.y = (unsigned int)f2bf(v.z) | ((unsigned int)f2bf(v.w) << 16);
  ((uint2*)Y)[i] = o;
}

// ---------------------------------------------------------------------------
// Unified bf16 GEMM: Y = X(bf16)[M x 2048] * W + bias, Wt[n][k] bf16.
// Async double-buffered LDS staging (GLOBAL_LOAD_ASYNC_TO_LDS_B128).
// mode 0: bf16 out scattered [b,h,s,d];  mode 1: bf16 out [b,h,d,s];
// mode 2: f32 out row-major [M x 2048].
// Block tile 128x128, 8 waves (4M x 2N), wave tile 32x64, BK=32.
// ---------------------------------------------------------------------------
__global__ __launch_bounds__(256)
void gemm_bf16_kernel(const unsigned short* __restrict__ X, const unsigned short* __restrict__ Wt,
                      const float* __restrict__ bias, unsigned short* __restrict__ OutB,
                      float* __restrict__ OutF, int mode) {
  __shared__ unsigned short As[2][128][32];
  __shared__ unsigned short Bs[2][128][32];
  const int tid = threadIdx.x, lane = tid & 31, wave = tid >> 5;
  const int half = (lane >> 4) & 1, l16 = lane & 15;
  const int mblk = blockIdx.y * 128, nblk = blockIdx.x * 128;
  const int wm = (wave >> 1) * 32, wn = (wave & 1) * 64;

  // per-thread async-copy slots: 2 b128 for As, 2 for Bs per tile
  const int e0 = tid * 8, e1 = (tid + 256) * 8;
  const int r0 = e0 >> 5, c0 = e0 & 31;
  const int r1 = e1 >> 5, c1 = e1 & 31;

  auto issue = [&](int buf, int k0) {
    copy16_g2l(X  + (size_t)(mblk + r0) * HIDDEN + k0 + c0, &As[buf][r0][c0]);
    copy16_g2l(X  + (size_t)(mblk + r1) * HIDDEN + k0 + c1, &As[buf][r1][c1]);
    copy16_g2l(Wt + (size_t)(nblk + r0) * HIDDEN + k0 + c0, &Bs[buf][r0][c0]);
    copy16_g2l(Wt + (size_t)(nblk + r1) * HIDDEN + k0 + c1, &Bs[buf][r1][c1]);
  };

  FragC acc[2][4];
#pragma unroll
  for (int mt = 0; mt < 2; ++mt)
#pragma unroll
    for (int nt = 0; nt < 4; ++nt)
#pragma unroll
      for (int r = 0; r < 8; ++r) acc[mt][nt].f[r] = 0.f;

  issue(0, 0);
  int buf = 0;
  for (int k0 = 0; k0 < HIDDEN; k0 += 32, buf ^= 1) {
    if (k0 + 32 < HIDDEN) {
      issue(buf ^ 1, k0 + 32);
      wait_async_keep4();        // current tile's 4 copies complete (in-order)
    } else {
      wait_async_all();
    }
    __syncthreads();             // all waves' LDS writes visible
    FragAB a[2], b[4];
#pragma unroll
    for (int mt = 0; mt < 2; ++mt) load_frag_a(a[mt], &As[buf][wm + mt * 16][0], 32, lane);
#pragma unroll
    for (int nt = 0; nt < 4; ++nt) load_frag_b(b[nt], &Bs[buf][wn + nt * 16][0], 32, lane);
#pragma unroll
    for (int mt = 0; mt < 2; ++mt)
#pragma unroll
      for (int nt = 0; nt < 4; ++nt)
        acc[mt][nt].v = wmma_bf16(a[mt].v, b[nt].v, acc[mt][nt].v);
    __syncthreads();             // done reading buf before it is refilled
  }

#pragma unroll
  for (int mt = 0; mt < 2; ++mt)
#pragma unroll
    for (int nt = 0; nt < 4; ++nt)
#pragma unroll
      for (int r = 0; r < 8; ++r) {
        const int gm = mblk + wm + mt * 16 + r + 8 * half;
        const int gn = nblk + wn + nt * 16 + l16;
        const float v = acc[mt][nt].f[r] + bias[gn];
        if (mode == 2) {
          OutF[(size_t)gm * HIDDEN + gn] = v;
        } else {
          const int b_ = gm >> 11, s = gm & (SEQ - 1);
          const int hh = gn >> 7, d = gn & (HEADDIM - 1);
          const size_t off = (mode == 0)
            ? ((((size_t)b_ * NHEADS + hh) * SEQ + s) * HEADDIM + d)
            : ((((size_t)b_ * NHEADS + hh) * HEADDIM + d) * SEQ + s);
          OutB[off] = f2bf(v);
        }
      }
}

// ---------------------------------------------------------------------------
// Attention: block = (q-tile of 16 rows, batch), 8 waves each own 256 keys.
// Pass1: online (m,l).  Pass2: P = exp(s-m)/l, mean accumulation, O = P*V.
// ---------------------------------------------------------------------------
__global__ __launch_bounds__(256)
void reward_attn_kernel(const unsigned short* __restrict__ Qb,
                        const unsigned short* __restrict__ Kb,
                        const unsigned short* __restrict__ Vt,
                        const float* __restrict__ rewards,
                        const unsigned char* __restrict__ amask,
                        const float* __restrict__ c01,
                        unsigned short* __restrict__ att,
                        float* __restrict__ meanw) {
  constexpr int TQ = 16;
  constexpr int WAVES = 8;
  constexpr int CHUNK = SEQ / WAVES;  // 256
  const float scale = 0.0883883476483184f;  // 1/sqrt(128)

  __shared__ unsigned char s_mask[SEQ];
  __shared__ float s_bias[TQ];
  __shared__ float s_m[WAVES][TQ];
  __shared__ float s_l[WAVES][TQ];
  __shared__ float s_M[TQ];
  __shared__ float s_Linv[TQ];
  __shared__ unsigned short s_p[WAVES][TQ][32];
  __shared__ float s_o4[4][TQ][HEADDIM];

  const int tid = threadIdx.x;
  const int lane = tid & 31;
  const int wave = tid >> 5;
  const int half = (lane >> 4) & 1;
  const int l16 = lane & 15;
  const int bb = blockIdx.y;
  const int qbase = blockIdx.x * TQ;
  const int kstart = wave * CHUNK;

  for (int i = tid; i < SEQ; i += 256) s_mask[i] = amask[(size_t)bb * SEQ + i];

  for (int h = 0; h < NHEADS; ++h) {
    if (tid < TQ)
      s_bias[tid] = rewards[(size_t)bb * SEQ + qbase + tid] * c01[h] + c01[NHEADS + h];
    __syncthreads();

    const unsigned short* Qh = Qb + (((size_t)bb * NHEADS + h) * SEQ + qbase) * HEADDIM;
    const unsigned short* Kh = Kb + ((size_t)bb * NHEADS + h) * (size_t)SEQ * HEADDIM;
    const unsigned short* Vh = Vt + ((size_t)bb * NHEADS + h) * (size_t)HEADDIM * SEQ;

    FragAB aq[4];
#pragma unroll
    for (int j = 0; j < 4; ++j) load_frag_a(aq[j], Qh + 32 * j, HEADDIM, lane);

    // ---- pass 1: per-row running max / sumexp over this wave's key chunk ----
    float mx[8], sm[8];
#pragma unroll
    for (int r = 0; r < 8; ++r) { mx[r] = -__builtin_inff(); sm[r] = 0.f; }

    for (int t = 0; t < CHUNK / 16; ++t) {
      const int keybase = kstart + t * 16;
      FragC acc;
#pragma unroll
      for (int r = 0; r < 8; ++r) acc.f[r] = 0.f;
#pragma unroll
      for (int j = 0; j < 4; ++j) {
        FragAB bk;
        load_frag_b(bk, Kh + (size_t)keybase * HEADDIM + 32 * j, HEADDIM, lane);
        acc.v = wmma_bf16(aq[j].v, bk.v, acc.v);
      }
      const int key = keybase + l16;
      const bool ok = s_mask[key] != 0;
#pragma unroll
      for (int r = 0; r < 8; ++r) {
        const int m = r + 8 * half;
        float s = acc.f[r] * scale + s_bias[m];
        s = ok ? s : -1e9f;
        float nm = fmaxf(mx[r], s);
        sm[r] = sm[r] * __expf(mx[r] - nm) + __expf(s - nm);
        mx[r] = nm;
      }
    }
#pragma unroll
    for (int d = 1; d < 16; d <<= 1) {
#pragma unroll
      for (int r = 0; r < 8; ++r) {
        float om = __shfl_xor(mx[r], d, 32);
        float ol = __shfl_xor(sm[r], d, 32);
        float nm = fmaxf(mx[r], om);
        sm[r] = sm[r] * __expf(mx[r] - nm) + ol * __expf(om - nm);
        mx[r] = nm;
      }
    }
    if (l16 == 0) {
#pragma unroll
      for (int r = 0; r < 8; ++r) {
        s_m[wave][r + 8 * half] = mx[r];
        s_l[wave][r + 8 * half] = sm[r];
      }
    }
    __syncthreads();
    if (tid < TQ) {
      float M = -__builtin_inff(), L = 0.f;
      for (int w = 0; w < WAVES; ++w) {
        float om = s_m[w][tid], ol = s_l[w][tid];
        float nm = fmaxf(M, om);
        L = L * __expf(M - nm) + ol * __expf(om - nm);
        M = nm;
      }
      s_M[tid] = M;
      s_Linv[tid] = (L > 0.f) ? (1.f / L) : 0.f;
    }
    __syncthreads();

    // ---- pass 2: P, mean, O = P*V ----
    FragC oacc[8];
#pragma unroll
    for (int dt = 0; dt < 8; ++dt)
#pragma unroll
      for (int r = 0; r < 8; ++r) oacc[dt].f[r] = 0.f;

    float Mrow[8], Linv[8];
#pragma unroll
    for (int r = 0; r < 8; ++r) { Mrow[r] = s_M[r + 8 * half]; Linv[r] = s_Linv[r + 8 * half]; }

    float* meanp = meanw + ((size_t)bb * SEQ + qbase) * SEQ;

    for (int tt = 0; tt < CHUNK / 32; ++tt) {
#pragma unroll
      for (int sub = 0; sub < 2; ++sub) {
        const int keybase = kstart + tt * 32 + sub * 16;
        FragC acc;
#pragma unroll
        for (int r = 0; r < 8; ++r) acc.f[r] = 0.f;
#pragma unroll
        for (int j = 0; j < 4; ++j) {
          FragAB bk;
          load_frag_b(bk, Kh + (size_t)keybase * HEADDIM + 32 * j, HEADDIM, lane);
          acc.v = wmma_bf16(aq[j].v, bk.v, acc.v);
        }
        const int key = keybase + l16;
        const bool ok = s_mask[key] != 0;
#pragma unroll
        for (int r = 0; r < 8; ++r) {
          const int m = r + 8 * half;
          float s = acc.f[r] * scale + s_bias[m];
          s = ok ? s : -1e9f;
          float p = __expf(s - Mrow[r]) * Linv[r];
          float* mp = meanp + (size_t)m * SEQ + key;
          float inc = p * (1.0f / NHEADS);
          if (h == 0) *mp = inc; else *mp += inc;   // block exclusively owns these rows
          s_p[wave][m][sub * 16 + l16] = f2bf(p);
        }
      }
      asm volatile("s_wait_dscnt 0" ::: "memory");  // same-wave LDS RAW before frag read
      FragAB ap;
      load_frag_a(ap, &s_p[wave][0][0], 32, lane);
      const int keyb = kstart + tt * 32;
#pragma unroll
      for (int dt = 0; dt < 8; ++dt) {
        FragAB bvf;
        load_frag_b(bvf, Vh + (size_t)(dt * 16) * SEQ + keyb, SEQ, lane);
        oacc[dt].v = wmma_bf16(ap.v, bvf.v, oacc[dt].v);
      }
    }

    // deterministic cross-wave O reduction (8 -> 4 buffers -> sum)
    if (wave < 4) {
#pragma unroll
      for (int dt = 0; dt < 8; ++dt)
#pragma unroll
        for (int r = 0; r < 8; ++r)
          s_o4[wave][r + 8 * half][dt * 16 + l16] = oacc[dt].f[r];
    }
    __syncthreads();
    if (wave >= 4) {
#pragma unroll
      for (int dt = 0; dt < 8; ++dt)
#pragma unroll
        for (int r = 0; r < 8; ++r)
          s_o4[wave - 4][r + 8 * half][dt * 16 + l16] += oacc[dt].f[r];
    }
    __syncthreads();
    const float* so = &s_o4[0][0][0];
    for (int e = tid; e < TQ * HEADDIM; e += 256) {
      float v = so[e] + so[TQ * HEADDIM + e] + so[2 * TQ * HEADDIM + e] + so[3 * TQ * HEADDIM + e];
      int row = e >> 7, col = e & 127;
      att[((size_t)bb * SEQ + qbase + row) * HIDDEN + h * HEADDIM + col] = f2bf(v);
    }
    __syncthreads();
  }
}

// ---------------------------------------------------------------------------
extern "C" void kernel_launch(void* const* d_in, const int* in_sizes, int n_in,
                              void* d_out, int out_size, void* d_ws, size_t ws_size,
                              hipStream_t stream) {
  (void)in_sizes; (void)n_in; (void)out_size; (void)ws_size;
  const float* query   = (const float*)d_in[0];
  const float* key     = (const float*)d_in[1];
  const float* value   = (const float*)d_in[2];
  const float* rewards = (const float*)d_in[3];
  const unsigned char* amask = (const unsigned char*)d_in[4];
  const float* Wq = (const float*)d_in[5];
  const float* bq = (const float*)d_in[6];
  const float* Wk = (const float*)d_in[7];
  const float* bk = (const float*)d_in[8];
  const float* Wv = (const float*)d_in[9];
  const float* bv = (const float*)d_in[10];
  const float* Wo = (const float*)d_in[11];
  const float* bo = (const float*)d_in[12];
  const float* Wr = (const float*)d_in[13];
  const float* br = (const float*)d_in[14];
  const float* Wa = (const float*)d_in[15];
  const float* ba = (const float*)d_in[16];

  char* ws = (char*)d_ws;
  const size_t wbytes = (size_t)HIDDEN * HIDDEN * sizeof(unsigned short);
  unsigned short* Wqt = (unsigned short*)ws; ws += wbytes;
  unsigned short* Wkt = (unsigned short*)ws; ws += wbytes;
  unsigned short* Wvt = (unsigned short*)ws; ws += wbytes;
  unsigned short* Wot = (unsigned short*)ws; ws += wbytes;
  const size_t abytes = (size_t)BATCH * SEQ * HIDDEN * sizeof(unsigned short);
  unsigned short* Qc  = (unsigned short*)ws; ws += abytes;   // bf16 copies of inputs
  unsigned short* Kc  = (unsigned short*)ws; ws += abytes;
  unsigned short* Vc  = (unsigned short*)ws; ws += abytes;
  unsigned short* Qb  = (unsigned short*)ws; ws += abytes;   // projected, per-head layouts
  unsigned short* Kb  = (unsigned short*)ws; ws += abytes;
  unsigned short* Vt  = (unsigned short*)ws; ws += abytes;
  unsigned short* att = (unsigned short*)ws; ws += abytes;
  float* c01 = (float*)ws; ws += 32 * sizeof(float);

  float* out   = (float*)d_out;
  float* meanw = out + (size_t)BATCH * SEQ * HIDDEN;

  reward_consts_kernel<<<NHEADS, 256, 0, stream>>>(Wr, br, Wa, ba, c01);

  dim3 tb(32, 8), tg(HIDDEN / 32, HIDDEN / 32);
  transpose_w_kernel<<<tg, tb, 0, stream>>>(Wq, Wqt);
  transpose_w_kernel<<<tg, tb, 0, stream>>>(Wk, Wkt);
  transpose_w_kernel<<<tg, tb, 0, stream>>>(Wv, Wvt);
  transpose_w_kernel<<<tg, tb, 0, stream>>>(Wo, Wot);

  const int cvt_blocks = (BATCH * SEQ * HIDDEN) / (256 * 4);
  cvt_bf16_kernel<<<cvt_blocks, 256, 0, stream>>>(query, Qc);
  cvt_bf16_kernel<<<cvt_blocks, 256, 0, stream>>>(key,   Kc);
  cvt_bf16_kernel<<<cvt_blocks, 256, 0, stream>>>(value, Vc);

  dim3 gg(HIDDEN / 128, (BATCH * SEQ) / 128);
  gemm_bf16_kernel<<<gg, 256, 0, stream>>>(Qc, Wqt, bq, Qb, nullptr, 0);
  gemm_bf16_kernel<<<gg, 256, 0, stream>>>(Kc, Wkt, bk, Kb, nullptr, 0);
  gemm_bf16_kernel<<<gg, 256, 0, stream>>>(Vc, Wvt, bv, Vt, nullptr, 1);

  dim3 ag(SEQ / 16, BATCH);
  reward_attn_kernel<<<ag, 256, 0, stream>>>(Qb, Kb, Vt, rewards, amask, c01, att, meanw);

  gemm_bf16_kernel<<<gg, 256, 0, stream>>>(att, Wot, bo, nullptr, out, 2);
}
